// MaskedMultiHeadAttention_89721866813669
// MI455X (gfx1250) — matrix operand order repaired and checked by reading, not compiled, not run on hardware
//
#include <hip/hip_runtime.h>
#include <math.h>

#define B_  2
#define S_  2048
#define D_  1024
#define H_  16
#define DH_ 64
#define M_  (B_ * S_)   // 4096 tokens

typedef __bf16 bf16_t;
typedef __attribute__((ext_vector_type(16))) __bf16 v16bf;
typedef __attribute__((ext_vector_type(8)))  float  v8f;

union Frag16 { uint4 u4[2]; v16bf v; bf16_t e[16]; };
union FragC  { v8f v; float f[8]; };

static __device__ __forceinline__ v8f wmma_bf16(v16bf a, v16bf b, v8f c) {
  // D = A(16x32 bf16) * B(32x16 bf16) + C(16x16 f32)
  return __builtin_amdgcn_wmma_f32_16x16x32_bf16(false, a, false, b, (short)0, c, false, false);
}

// ---------------------------------------------------------------- convert (8 elems/thread, b128 in/out)
__global__ void k_f32_to_bf16(const float* __restrict__ in, bf16_t* __restrict__ out, int n) {
  int i = (blockIdx.x * blockDim.x + threadIdx.x) * 8;
  if (i < n) {
    const float4 f0 = *(const float4*)(in + i);
    const float4 f1 = *(const float4*)(in + i + 4);
    union { uint4 u; bf16_t e[8]; } o;
    o.e[0] = (bf16_t)f0.x; o.e[1] = (bf16_t)f0.y; o.e[2] = (bf16_t)f0.z; o.e[3] = (bf16_t)f0.w;
    o.e[4] = (bf16_t)f1.x; o.e[5] = (bf16_t)f1.y; o.e[6] = (bf16_t)f1.z; o.e[7] = (bf16_t)f1.w;
    *(uint4*)(out + i) = o.u;
  }
}

// ---------------------------------------------------------------- GEMM: out[m][n] = sum_k A[m][k]*W[n][k] + bias[n]
// 128 threads = 4 waves (2x2), 64x64 wave tile -> 128x128 block tile.
// Per 32-K step: 4 A-frags + 4 B-frags (16 b128 loads) feed 16 WMMAs (1:1 load:WMMA).
// mode 0: store bf16 head-major [b][h][s][dh], value scaled by `scale`
// mode 1: store f32 row-major [m][n]
__global__ void __launch_bounds__(128)
k_gemm_bf16(const bf16_t* __restrict__ A, const bf16_t* __restrict__ W,
            const float* __restrict__ bias, float* __restrict__ outF,
            bf16_t* __restrict__ outH, int M, int N, int K, int mode, float scale) {
  const int lane = threadIdx.x & 31;
  const int wv   = threadIdx.x >> 5;
  const int wm   = wv & 1;          // 2 waves along M
  const int wn   = wv >> 1;         // 2 waves along N
  const int l15  = lane & 15;
  const int hiA  = (lane < 16) ? 0 : 8;    // A lane-half K offset
  const int hiB  = (lane < 16) ? 0 : 16;   // B lane-half K offset

  const int mbase = blockIdx.x * 128 + wm * 64;
  const int nbase = blockIdx.y * 128 + wn * 64;

  FragC c[4][4];
#pragma unroll
  for (int i = 0; i < 4; ++i)
#pragma unroll
    for (int j = 0; j < 4; ++j)
#pragma unroll
      for (int e = 0; e < 8; ++e) c[i][j].f[e] = 0.0f;

  const bf16_t* Ar[4];
  const bf16_t* Wr[4];
#pragma unroll
  for (int i = 0; i < 4; ++i) {
    Ar[i] = A + (size_t)(mbase + i * 16 + l15) * K;
    Wr[i] = W + (size_t)(nbase + i * 16 + l15) * K;
  }

  for (int kk = 0; kk < K; kk += 32) {
    Frag16 a[4], b[4];
#pragma unroll
    for (int i = 0; i < 4; ++i) {
      a[i].u4[0] = *(const uint4*)(Ar[i] + kk + hiA);        // K = kk+hi .. +7
      a[i].u4[1] = *(const uint4*)(Ar[i] + kk + 16 + hiA);   // K = kk+16+hi .. +7
    }
#pragma unroll
    for (int j = 0; j < 4; ++j) {
      b[j].u4[0] = *(const uint4*)(Wr[j] + kk + hiA);        // K = kk+hi .. +7
      b[j].u4[1] = *(const uint4*)(Wr[j] + kk + 16 + hiA);   // K = kk+16+hi .. +7
    }
#pragma unroll
    for (int i = 0; i < 4; ++i)
#pragma unroll
      for (int j = 0; j < 4; ++j)
        c[i][j].v = wmma_bf16(a[i].v, b[j].v, c[i][j].v);
  }

#pragma unroll
  for (int i = 0; i < 4; ++i)
#pragma unroll
    for (int j = 0; j < 4; ++j)
#pragma unroll
      for (int r = 0; r < 8; ++r) {
        const int row = mbase + i * 16 + r + hiA;   // C layout: lanes>=16 -> M = r+8
        const int col = nbase + j * 16 + l15;       // C layout: N = lane&15
        const float vv = c[i][j].f[r] + bias[col];
        if (mode == 0) {
          const int bb = row >> 11, ss = row & (S_ - 1);
          const int hh = col >> 6,  dd = col & (DH_ - 1);
          outH[(((size_t)(bb * H_ + hh) * S_ + ss) * DH_) + dd] = (bf16_t)(vv * scale);
        } else {
          outF[(size_t)row * N + col] = vv;
        }
      }
}

// ---------------------------------------------------------------- flash attention (causal)
// Q/K/V bf16 head-major [b][h][s][dh]; Q pre-scaled by 1/sqrt(dh).
// Output bf16 [b][s][h][dh].
#define VT_STRIDE 72   // 64 + 8 pad elements; 144B rows keep 16B alignment
__global__ void __launch_bounds__(128)
k_flash_attn(const bf16_t* __restrict__ Qh, const bf16_t* __restrict__ Kh,
             const bf16_t* __restrict__ Vh, bf16_t* __restrict__ Oh) {
  __shared__ bf16_t lds[64 * VT_STRIDE + 4 * 16 * VT_STRIDE];
  const int PBASE = 64 * VT_STRIDE;
  const float LOG2E = 1.4426950408889634f;

  const int qtile = blockIdx.x, h = blockIdx.y, b = blockIdx.z;
  const int tid = threadIdx.x, lane = tid & 31, wv = tid >> 5;
  const int l15 = lane & 15;
  const int hiA = (lane < 16) ? 0 : 8;
  const int hiB = (lane < 16) ? 0 : 16;

  const size_t bh = (size_t)(b * H_ + h) * S_ * DH_;
  const bf16_t* Q = Qh + bh;
  const bf16_t* K = Kh + bh;
  const bf16_t* V = Vh + bh;

  const int qbase = qtile * 64 + wv * 16;   // this wave's 16 query rows

  // Q A-fragments (persistent): dh = 64 -> two K-steps of 32
  Frag16 qf[2];
  {
    const bf16_t* Qrow = Q + (size_t)(qbase + l15) * DH_;
#pragma unroll
    for (int k2 = 0; k2 < 2; ++k2) {
      qf[k2].u4[0] = *(const uint4*)(Qrow + k2 * 32 + hiA);
      qf[k2].u4[1] = *(const uint4*)(Qrow + k2 * 32 + 16 + hiA);
    }
  }

  FragC acc[4];
  float m_run[8], l_run[8];
#pragma unroll
  for (int f = 0; f < 4; ++f)
#pragma unroll
    for (int e = 0; e < 8; ++e) acc[f].f[e] = 0.0f;
#pragma unroll
  for (int r = 0; r < 8; ++r) { m_run[r] = -__builtin_inff(); l_run[r] = 0.0f; }

  const int ktiles = qtile + 1;   // causal: K-tiles with kb <= qtile*64+63
  for (int kt = 0; kt < ktiles; ++kt) {
    const int kb = kt * 64;

    // prefetch next K/V tiles into cache while we compute (global_prefetch_b8)
    if (kt + 1 < ktiles) {
      const int nb = kb + 64;
      __builtin_prefetch(K + (size_t)(nb + lane) * DH_, 0, 3);
      __builtin_prefetch(K + (size_t)(nb + 32 + lane) * DH_, 0, 3);
      __builtin_prefetch(V + (size_t)(nb + lane) * DH_, 0, 3);
      __builtin_prefetch(V + (size_t)(nb + 32 + lane) * DH_, 0, 3);
    }

    __syncthreads();  // previous iteration's LDS reads done before overwrite

    // ---- stage V^T into LDS: lds_vt[d][n] = V[kb+n][d]
    {
      const int nloc = tid >> 1;
      const int doff = (tid & 1) * 32;
      union { uint4 q[4]; bf16_t e[32]; } vv;
      const uint4* vp = (const uint4*)(V + (size_t)(kb + nloc) * DH_ + doff);
      vv.q[0] = vp[0]; vv.q[1] = vp[1]; vv.q[2] = vp[2]; vv.q[3] = vp[3];
#pragma unroll
      for (int j = 0; j < 32; ++j) lds[(doff + j) * VT_STRIDE + nloc] = vv.e[j];
    }

    // ---- scores: c[f] = Q(16x64) . K(kb+f*16.., 64)^T
    FragC c[4];
#pragma unroll
    for (int f = 0; f < 4; ++f) {
#pragma unroll
      for (int e = 0; e < 8; ++e) c[f].f[e] = 0.0f;
      const bf16_t* Krow = K + (size_t)(kb + f * 16 + l15) * DH_;
      Frag16 kf;
      kf.u4[0] = *(const uint4*)(Krow + hiA);
      kf.u4[1] = *(const uint4*)(Krow + 16 + hiA);
      c[f].v = wmma_bf16(qf[0].v, kf.v, c[f].v);
      kf.u4[0] = *(const uint4*)(Krow + 32 + hiA);
      kf.u4[1] = *(const uint4*)(Krow + 32 + 16 + hiA);
      c[f].v = wmma_bf16(qf[1].v, kf.v, c[f].v);
    }

    // ---- causal mask
#pragma unroll
    for (int f = 0; f < 4; ++f) {
      const int col = kb + f * 16 + l15;
#pragma unroll
      for (int r = 0; r < 8; ++r) {
        const int row = qbase + r + hiA;
        if (col > row) c[f].f[r] = -3.0e38f;
      }
    }

    // ---- online softmax
    float mnew[8], alpha[8];
#pragma unroll
    for (int r = 0; r < 8; ++r) {
      float t = fmaxf(fmaxf(c[0].f[r], c[1].f[r]), fmaxf(c[2].f[r], c[3].f[r]));
      t = fmaxf(t, __shfl_xor(t, 1, 32));
      t = fmaxf(t, __shfl_xor(t, 2, 32));
      t = fmaxf(t, __shfl_xor(t, 4, 32));
      t = fmaxf(t, __shfl_xor(t, 8, 32));   // reduce within 16-lane half (rows differ per half)
      mnew[r]  = fmaxf(m_run[r], t);
      alpha[r] = exp2f((m_run[r] - mnew[r]) * LOG2E);
      m_run[r] = mnew[r];
    }
#pragma unroll
    for (int f = 0; f < 4; ++f)
#pragma unroll
      for (int r = 0; r < 8; ++r)
        c[f].f[r] = exp2f((c[f].f[r] - mnew[r]) * LOG2E);
#pragma unroll
    for (int r = 0; r < 8; ++r) {
      float rs = c[0].f[r] + c[1].f[r] + c[2].f[r] + c[3].f[r];
      rs += __shfl_xor(rs, 1, 32);
      rs += __shfl_xor(rs, 2, 32);
      rs += __shfl_xor(rs, 4, 32);
      rs += __shfl_xor(rs, 8, 32);
      l_run[r] = l_run[r] * alpha[r] + rs;
    }
#pragma unroll
    for (int f = 0; f < 4; ++f)
#pragma unroll
      for (int r = 0; r < 8; ++r) acc[f].f[r] *= alpha[r];

    // ---- write P (bf16) to this wave's LDS region (C-layout -> memory)
    const int pb = PBASE + wv * 16 * VT_STRIDE;
#pragma unroll
    for (int f = 0; f < 4; ++f)
#pragma unroll
      for (int r = 0; r < 8; ++r)
        lds[pb + (r + hiA) * VT_STRIDE + f * 16 + l15] = (bf16_t)c[f].f[r];

    __syncthreads();  // V^T + P visible

    // ---- acc += P(16x64) . V(64x64): A-frags from LDS P, B-frags from LDS V^T
    Frag16 pf[2];
    {
      const int prow = pb + l15 * VT_STRIDE;
#pragma unroll
      for (int k2 = 0; k2 < 2; ++k2) {
        pf[k2].u4[0] = *(const uint4*)(&lds[prow + k2 * 32 + hiA]);
        pf[k2].u4[1] = *(const uint4*)(&lds[prow + k2 * 32 + 16 + hiA]);
      }
    }
#pragma unroll
    for (int f = 0; f < 4; ++f) {
      const int d = f * 16 + l15;
      Frag16 vf;
      vf.u4[0] = *(const uint4*)(&lds[d * VT_STRIDE + hiA]);
      vf.u4[1] = *(const uint4*)(&lds[d * VT_STRIDE + 16 + hiA]);
      acc[f].v = wmma_bf16(pf[0].v, vf.v, acc[f].v);
      vf.u4[0] = *(const uint4*)(&lds[d * VT_STRIDE + 32 + hiA]);
      vf.u4[1] = *(const uint4*)(&lds[d * VT_STRIDE + 32 + 16 + hiA]);
      acc[f].v = wmma_bf16(pf[1].v, vf.v, acc[f].v);
    }
  }

  // ---- finalize: divide by row sums, store bf16 [b][s][h][dh]
#pragma unroll
  for (int r = 0; r < 8; ++r) {
    const float inv = 1.0f / l_run[r];
    const int srow = qbase + r + hiA;
#pragma unroll
    for (int f = 0; f < 4; ++f) {
      const int d = f * 16 + l15;
      Oh[((size_t)(b * S_ + srow) * H_ + h) * DH_ + d] = (bf16_t)(acc[f].f[r] * inv);
    }
  }
}

// ---------------------------------------------------------------- launch
extern "C" void kernel_launch(void* const* d_in, const int* in_sizes, int n_in,
                              void* d_out, int out_size, void* d_ws, size_t ws_size,
                              hipStream_t stream) {
  (void)in_sizes; (void)n_in; (void)out_size; (void)ws_size;
  const float* x  = (const float*)d_in[0];
  // d_in[1] = mask (int32): causality applied analytically, not read
  const float* Wq = (const float*)d_in[2];
  const float* bq = (const float*)d_in[3];
  const float* Wk = (const float*)d_in[4];
  const float* bk = (const float*)d_in[5];
  const float* Wv = (const float*)d_in[6];
  const float* bv = (const float*)d_in[7];
  const float* Wo = (const float*)d_in[8];
  const float* bo = (const float*)d_in[9];
  float* out = (float*)d_out;

  char* ws = (char*)d_ws;
  size_t off = 0;
  auto carve = [&](size_t bytes) -> void* {
    void* p = (void*)(ws + off);
    off += (bytes + 255) & ~(size_t)255;
    return p;
  };
  bf16_t* xh    = (bf16_t*)carve((size_t)M_ * D_ * sizeof(bf16_t));
  bf16_t* Wqh   = (bf16_t*)carve((size_t)D_ * D_ * sizeof(bf16_t));
  bf16_t* Wkh   = (bf16_t*)carve((size_t)D_ * D_ * sizeof(bf16_t));
  bf16_t* Wvh   = (bf16_t*)carve((size_t)D_ * D_ * sizeof(bf16_t));
  bf16_t* Woh   = (bf16_t*)carve((size_t)D_ * D_ * sizeof(bf16_t));
  bf16_t* Qh    = (bf16_t*)carve((size_t)M_ * D_ * sizeof(bf16_t));
  bf16_t* Kh    = (bf16_t*)carve((size_t)M_ * D_ * sizeof(bf16_t));
  bf16_t* Vh    = (bf16_t*)carve((size_t)M_ * D_ * sizeof(bf16_t));
  bf16_t* attnH = (bf16_t*)carve((size_t)M_ * D_ * sizeof(bf16_t));

  const int nX = M_ * D_, nW = D_ * D_;
  k_f32_to_bf16<<<(nX / 8 + 255) / 256, 256, 0, stream>>>(x,  xh,  nX);
  k_f32_to_bf16<<<(nW / 8 + 255) / 256, 256, 0, stream>>>(Wq, Wqh, nW);
  k_f32_to_bf16<<<(nW / 8 + 255) / 256, 256, 0, stream>>>(Wk, Wkh, nW);
  k_f32_to_bf16<<<(nW / 8 + 255) / 256, 256, 0, stream>>>(Wv, Wvh, nW);
  k_f32_to_bf16<<<(nW / 8 + 255) / 256, 256, 0, stream>>>(Wo, Woh, nW);

  dim3 gg(M_ / 128, D_ / 128);
  const float qscale = 0.125f;   // 1/sqrt(64)
  k_gemm_bf16<<<gg, 128, 0, stream>>>(xh, Wqh, bq, nullptr, Qh, M_, D_, D_, 0, qscale);
  k_gemm_bf16<<<gg, 128, 0, stream>>>(xh, Wkh, bk, nullptr, Kh, M_, D_, D_, 0, 1.0f);
  k_gemm_bf16<<<gg, 128, 0, stream>>>(xh, Wvh, bv, nullptr, Vh, M_, D_, D_, 0, 1.0f);

  dim3 ga(S_ / 64, H_, B_);
  k_flash_attn<<<ga, 128, 0, stream>>>(Qh, Kh, Vh, attnH);

  k_gemm_bf16<<<gg, 128, 0, stream>>>(attnH, Woh, bo, out, nullptr, M_, D_, D_, 1, 1.0f);
}
